// Block_2439541424589
// MI455X (gfx1250) — matrix operand order repaired and checked by reading, not compiled, not run on hardware
//
#include <hip/hip_runtime.h>

// ---------------------------------------------------------------------------
// Types for CDNA5 WMMA (gfx1250, wave32)
// ---------------------------------------------------------------------------
typedef __bf16 bf16;
typedef __attribute__((ext_vector_type(16))) __bf16 v16bf;
typedef __attribute__((ext_vector_type(8)))  __bf16 v8bf;
typedef __attribute__((ext_vector_type(8)))  float  v8f;
typedef __attribute__((ext_vector_type(4)))  unsigned int u32x4;
typedef __attribute__((ext_vector_type(8)))  int i32x8;
typedef __attribute__((ext_vector_type(4)))  int i32x4;

#define DIM    768
#define HIDDEN 3072
#define NCTX   1024
#define BATCH  8
#define HEADS  12
#define HDIM   64
#define ROWS   (BATCH * NCTX)          // 8192
#define QKVN   (3 * DIM)               // 2304

// Tensor Data Mover availability (device pass only; host pass parses fallback)
#if defined(__gfx1250__) && __has_builtin(__builtin_amdgcn_tensor_load_to_lds) && \
    __has_builtin(__builtin_amdgcn_s_wait_tensorcnt)
#define USE_TDM 1
#else
#define USE_TDM 0
#endif

__device__ __forceinline__ v8f wmma_bf16(v16bf a, v16bf b, v8f c) {
  // D = A(16x32 bf16) x B(32x16 bf16) + C(16x16 f32)
  return __builtin_amdgcn_wmma_f32_16x16x32_bf16(false, a, false, b,
                                                 (short)0, c, false, false);
}

// A-operand fragment: lane holds row (lane&15); 8 bf16 at k0=(lane>>4)*16 and
// 8 bf16 at k0+8 (contiguous k = 16h + s, same order as the B loads).
__device__ __forceinline__ v16bf ld_frag_a(const bf16* lo, const bf16* hi) {
  v8bf a = *(const v8bf*)lo;
  v8bf b = *(const v8bf*)hi;
  v16bf r;
#pragma unroll
  for (int i = 0; i < 8; ++i) { r[i] = a[i]; r[i + 8] = b[i]; }
  return r;
}

// ---------------------------------------------------------------------------
// 16-lane XOR butterfly: v_permlane16_b32 (VALU) when available, else shfl
// (which lowers to ds_bpermute). XOR patterns are self-inverse, so gather
// direction is irrelevant.
// ---------------------------------------------------------------------------
#if __has_builtin(__builtin_amdgcn_permlane16)
__device__ __forceinline__ float xor16f(float v, unsigned s0, unsigned s1) {
  unsigned u = __float_as_uint(v);
  return __uint_as_float(__builtin_amdgcn_permlane16(u, u, s0, s1, false, false));
}
#define XOR16_1(v) xor16f((v), 0x67452301u, 0xEFCDAB89u)
#define XOR16_2(v) xor16f((v), 0x54761032u, 0xDCFE98BAu)
#define XOR16_4(v) xor16f((v), 0x32107654u, 0xBA98FEDCu)
#define XOR16_8(v) xor16f((v), 0xFEDCBA98u, 0x76543210u)
#else
#define XOR16_1(v) __shfl_xor((v), 1, 16)
#define XOR16_2(v) __shfl_xor((v), 2, 16)
#define XOR16_4(v) __shfl_xor((v), 4, 16)
#define XOR16_8(v) __shfl_xor((v), 8, 16)
#endif

__device__ __forceinline__ float redmax16(float v) {
  v = fmaxf(v, XOR16_8(v)); v = fmaxf(v, XOR16_4(v));
  v = fmaxf(v, XOR16_2(v)); v = fmaxf(v, XOR16_1(v));
  return v;
}
__device__ __forceinline__ float redsum16(float v) {
  v += XOR16_8(v); v += XOR16_4(v); v += XOR16_2(v); v += XOR16_1(v);
  return v;
}

// ---------------------------------------------------------------------------
// TDM: async 2D tile load (128 rows x 32 bf16) global -> LDS with HW padding
// of 4 DWORDs after every 16 DWORDs => 80B LDS row pitch (matches [*][40]).
// D# layout per CDNA5 ISA.
// ---------------------------------------------------------------------------
#if USE_TDM
__device__ __forceinline__ unsigned lds_lo(const void* p) {
  // LDS flat aperture: low 32 bits are the wave-relative LDS byte offset
  return (unsigned)(uintptr_t)p;
}
__device__ __forceinline__ void tdm_load_tile(unsigned lds_off, const bf16* g, int K) {
  const unsigned long long ga = (unsigned long long)(uintptr_t)g;
  u32x4 g0;
  g0[0] = 1u;                                   // count=1, user mode, no gather
  g0[1] = lds_off;                              // LDS byte address
  g0[2] = (unsigned)ga;                         // global addr [31:0]
  g0[3] = (unsigned)((ga >> 32) & 0x01FFFFFFu)  // global addr [56:32]
          | (2u << 30);                         // type = 2 ("image")
  i32x8 g1;
  g1[0] = (int)((1u << 16)      // data_size = 1 -> 2 bytes
              | (1u << 20)      // pad_enable
              | (3u << 22)      // pad_interval: 16 DWORDs (tile row = 64B)
              | (3u << 25));    // pad_amount: 4 DWORDs (16B) -> 80B pitch
  g1[1] = (int)0xFFFF0000u;                     // tensor_dim0 lo16 (huge: no OOB)
  g1[2] = (int)0xFFFF7FFFu;                     // tensor_dim0 hi16 | tensor_dim1 lo16
  g1[3] = (int)((32u << 16) | 0x7FFFu);         // tensor_dim1 hi16 | tile_dim0=32
  g1[4] = 128;                                  // tile_dim1=128 (tile_dim2=0)
  g1[5] = K;                                    // tensor_dim0_stride [31:0] (elems)
  g1[6] = 0;                                    // stride hi / dim1_stride lo
  g1[7] = 0;
  const i32x4 z4 = {0, 0, 0, 0};
#if __has_include(<hip/amd_detail/amd_gfx1250_TDM.h>)
  const i32x8 z8 = {0, 0, 0, 0, 0, 0, 0, 0};
  __builtin_amdgcn_tensor_load_to_lds(g0, g1, z4, z4, z8, 0);
#else
  __builtin_amdgcn_tensor_load_to_lds(g0, g1, z4, z4, 0);
#endif
}
#endif  // USE_TDM

// ---------------------------------------------------------------------------
// fp32 -> bf16 conversion (weights)
// ---------------------------------------------------------------------------
__global__ __launch_bounds__(256) void cvt_bf16_kernel(const float* __restrict__ s,
                                                       bf16* __restrict__ d, int n) {
  int i = blockIdx.x * 256 + threadIdx.x;
  if (i < n) d[i] = (bf16)s[i];
}

// ---------------------------------------------------------------------------
// LayerNorm over DIM=768, one row per block (256 threads, 3 elems/thread)
// ---------------------------------------------------------------------------
__global__ __launch_bounds__(256) void ln_kernel(const float* __restrict__ x,
                                                 const float* __restrict__ g,
                                                 const float* __restrict__ bta,
                                                 bf16* __restrict__ out) {
  const int row = blockIdx.x;
  const int t = threadIdx.x;
  const float* xr = x + (size_t)row * DIM;
  float v0 = xr[t], v1 = xr[t + 256], v2 = xr[t + 512];

  __shared__ float red[256];
  red[t] = v0 + v1 + v2;
  __syncthreads();
#pragma unroll
  for (int off = 128; off > 0; off >>= 1) {
    if (t < off) red[t] += red[t + off];
    __syncthreads();
  }
  const float mu = red[0] * (1.0f / DIM);
  __syncthreads();

  const float d0 = v0 - mu, d1 = v1 - mu, d2 = v2 - mu;
  red[t] = d0 * d0 + d1 * d1 + d2 * d2;
  __syncthreads();
#pragma unroll
  for (int off = 128; off > 0; off >>= 1) {
    if (t < off) red[t] += red[t + off];
    __syncthreads();
  }
  const float rstd = rsqrtf(red[0] * (1.0f / DIM) + 1e-5f);

  bf16* orow = out + (size_t)row * DIM;
  orow[t]       = (bf16)(d0 * rstd * g[t]       + bta[t]);
  orow[t + 256] = (bf16)(d1 * rstd * g[t + 256] + bta[t + 256]);
  orow[t + 512] = (bf16)(d2 * rstd * g[t + 512] + bta[t + 512]);
}

// ---------------------------------------------------------------------------
// One 128x128x32 compute step from LDS tiles (8 waves, 8 WMMA each)
// ---------------------------------------------------------------------------
__device__ __forceinline__ void gemm_step(const bf16 (*Asb)[40], const bf16 (*Bsb)[40],
                                          int wr, int wc, int lrow, int lhi,
                                          v8f acc[2][4]) {
  v16bf af[2], bfr[4];
#pragma unroll
  for (int i = 0; i < 2; ++i) {
    const int r = wr * 32 + i * 16 + lrow;
    const int kk = lhi * 16;
    af[i] = ld_frag_a(&Asb[r][kk], &Asb[r][kk + 8]);
  }
#pragma unroll
  for (int j = 0; j < 4; ++j) {
    const int c = wc * 64 + j * 16 + lrow;
    bfr[j] = *(const v16bf*)(&Bsb[c][lhi * 16]);   // 32B contiguous B-operand
  }
#pragma unroll
  for (int i = 0; i < 2; ++i)
#pragma unroll
    for (int j = 0; j < 4; ++j)
      acc[i][j] = wmma_bf16(af[i], bfr[j], acc[i][j]);
}

// ---------------------------------------------------------------------------
// Tiled WMMA GEMM:  Out[M,N] = A[M,K](bf16) * W[N,K]^T(bf16)  (+ epilogue)
//   EPI 0: store bf16 | EPI 1: fp32 = acc + bias + res | EPI 2: bf16 gelu(acc+bias)
// Block 256 thr = 8 waves; tile 128x128, BK=32, TDM double-buffered LDS staging.
// ---------------------------------------------------------------------------
template <int EPI>
__global__ __launch_bounds__(256) void gemm_bf16_kernel(
    const bf16* __restrict__ A, const bf16* __restrict__ W,
    int M, int N, int K,
    const float* __restrict__ bias, const float* __restrict__ res,
    float* __restrict__ outF, bf16* __restrict__ outB) {
  __shared__ bf16 As[2][128][40];   // 80B pitch: 16B aligned, conflict-free
  __shared__ bf16 Bs[2][128][40];

  const int tid  = threadIdx.x;
  const int wave = tid >> 5;
  const int lane = tid & 31;
  const int lrow = lane & 15;
  const int lhi  = lane >> 4;
  const int wr   = wave & 3;          // 4 row-waves
  const int wc   = wave >> 2;         // 2 col-waves
  const int m0   = blockIdx.y * 128;
  const int n0   = blockIdx.x * 128;

  v8f acc[2][4];
#pragma unroll
  for (int i = 0; i < 2; ++i)
#pragma unroll
    for (int j = 0; j < 4; ++j)
#pragma unroll
      for (int r = 0; r < 8; ++r) acc[i][j][r] = 0.0f;

  const int nIter = K >> 5;

#if USE_TDM
  if (wave == 0) {
    tdm_load_tile(lds_lo(&As[0][0][0]), A + (size_t)m0 * K, K);
    tdm_load_tile(lds_lo(&Bs[0][0][0]), W + (size_t)n0 * K, K);
  }
  for (int it = 0; it < nIter; ++it) {
    const int buf = it & 1;
    if (wave == 0) {
      if (it + 1 < nIter) {
        const int k1 = (it + 1) << 5;
        tdm_load_tile(lds_lo(&As[buf ^ 1][0][0]), A + (size_t)m0 * K + k1, K);
        tdm_load_tile(lds_lo(&Bs[buf ^ 1][0][0]), W + (size_t)n0 * K + k1, K);
        __builtin_amdgcn_s_wait_tensorcnt((short)2);   // tile `it` complete
      } else {
        __builtin_amdgcn_s_wait_tensorcnt((short)0);
      }
    }
    __syncthreads();
    gemm_step(As[buf], Bs[buf], wr, wc, lrow, lhi, acc);
    __syncthreads();   // all reads of `buf` done before it is reloaded
  }
#else
  const int ldr = tid >> 1;           // 0..127
  const int ldc = (tid & 1) * 16;     // 0 or 16 (32B chunk)
  const bf16* Ag = A + (size_t)(m0 + ldr) * K + ldc;
  const bf16* Wg = W + (size_t)(n0 + ldr) * K + ldc;
  for (int it = 0; it < nIter; ++it) {
    const int k0 = it << 5;
    *(v16bf*)(&As[0][ldr][ldc]) = *(const v16bf*)(Ag + k0);
    *(v16bf*)(&Bs[0][ldr][ldc]) = *(const v16bf*)(Wg + k0);
    __syncthreads();
    gemm_step(As[0], Bs[0], wr, wc, lrow, lhi, acc);
    __syncthreads();
  }
#endif

  // Epilogue: C layout -> row = base + r + 8*(lane>>4), col = base + (lane&15)
#pragma unroll
  for (int i = 0; i < 2; ++i) {
#pragma unroll
    for (int j = 0; j < 4; ++j) {
#pragma unroll
      for (int r = 0; r < 8; ++r) {
        const int row = m0 + wr * 32 + i * 16 + r + lhi * 8;
        const int col = n0 + wc * 64 + j * 16 + lrow;
        float v = acc[i][j][r];
        const size_t idx = (size_t)row * N + col;
        if (EPI == 0) {
          outB[idx] = (bf16)v;
        } else if (EPI == 1) {
          outF[idx] = v + bias[col] + res[idx];
        } else {  // exact GELU
          v += bias[col];
          v = 0.5f * v * (1.0f + erff(v * 0.70710678118654752f));
          outB[idx] = (bf16)v;
        }
      }
    }
  }
}

// ---------------------------------------------------------------------------
// Split QKV [ROWS, 2304] -> q[BH,N,64], k[BH,N,64], vT[BH,64,N]
// ---------------------------------------------------------------------------
__global__ __launch_bounds__(256) void qkv_split_kernel(const bf16* __restrict__ qkv,
                                                        bf16* __restrict__ q,
                                                        bf16* __restrict__ k,
                                                        bf16* __restrict__ vt) {
  const size_t idx = (size_t)blockIdx.x * 256 + threadIdx.x;  // < ROWS*QKVN
  const int m = (int)(idx / QKVN);
  const int c = (int)(idx % QKVN);
  const int sel = c / DIM;
  const int rr  = c % DIM;
  const int h = rr >> 6, d = rr & 63;
  const int b = m >> 10, n = m & 1023;
  const size_t bh = (size_t)b * HEADS + h;
  const bf16 v = qkv[idx];
  if (sel == 0)       q [(bh * NCTX + n) * HDIM + d] = v;
  else if (sel == 1)  k [(bh * NCTX + n) * HDIM + d] = v;
  else                vt[(bh * HDIM + d) * NCTX + n] = v;
}

// ---------------------------------------------------------------------------
// Flash attention: grid = BH * (N/128); block = 256 (8 waves, 16 q-rows each).
// S = Q*K^T (WMMA), online softmax (permlane16 butterflies), P restaged via
// wave-private LDS (C-layout -> A-layout), O += P*V^T (WMMA).
// ---------------------------------------------------------------------------
__global__ __launch_bounds__(256) void attn_kernel(const bf16* __restrict__ Q,
                                                   const bf16* __restrict__ Kt,
                                                   const bf16* __restrict__ VT,
                                                   bf16* __restrict__ ctx) {
  __shared__ bf16 plds[8][16][40];     // per-wave P tile, padded pitch

  const int tid  = threadIdx.x;
  const int wave = tid >> 5;
  const int lane = tid & 31;
  const int lrow = lane & 15;
  const int lhi  = lane >> 4;
  const int bh   = blockIdx.x >> 3;                 // 0..95
  const int rblk = (blockIdx.x & 7) * 128 + wave * 16;
  const int b    = bh / HEADS, h = bh % HEADS;

  const bf16* qb = Q  + (size_t)bh * NCTX * HDIM;
  const bf16* kb = Kt + (size_t)bh * NCTX * HDIM;
  const bf16* vb = VT + (size_t)bh * HDIM * NCTX;

  v16bf aq[2];
  {
    const int r = rblk + lrow;
#pragma unroll
    for (int kc = 0; kc < 2; ++kc) {
      const bf16* base = qb + (size_t)r * HDIM + kc * 32 + lhi * 16;
      aq[kc] = ld_frag_a(base, base + 8);
    }
  }

  v8f O[4];
  float mrun[8], lsum[8];
#pragma unroll
  for (int t = 0; t < 4; ++t)
#pragma unroll
    for (int r = 0; r < 8; ++r) O[t][r] = 0.0f;
#pragma unroll
  for (int r = 0; r < 8; ++r) { mrun[r] = -3.0e38f; lsum[r] = 0.0f; }

  const float scale = 0.125f;  // 64^-0.5

  for (int j2 = 0; j2 < NCTX / 32; ++j2) {
    // prefetch next K/V chunk while this one computes
    if (j2 + 1 < NCTX / 32) {
      __builtin_prefetch((const void*)(kb + (size_t)(j2 * 32 + 32 + lane) * HDIM), 0, 0);
      __builtin_prefetch((const void*)(vb + (size_t)lane * NCTX + j2 * 32 + 32), 0, 0);
      __builtin_prefetch((const void*)(vb + (size_t)(lane + 32) * NCTX + j2 * 32 + 32), 0, 0);
    }

    v8f s[2];
#pragma unroll
    for (int jc = 0; jc < 2; ++jc) {
      const int kr = j2 * 32 + jc * 16 + lrow;
      const bf16* krow = kb + (size_t)kr * HDIM;
      v16bf b0 = *(const v16bf*)(krow + lhi * 16);        // d = 0..31
      v16bf b1 = *(const v16bf*)(krow + 32 + lhi * 16);   // d = 32..63
      v8f t;
#pragma unroll
      for (int r = 0; r < 8; ++r) t[r] = 0.0f;
      t = wmma_bf16(aq[0], b0, t);
      t = wmma_bf16(aq[1], b1, t);
#pragma unroll
      for (int r = 0; r < 8; ++r) s[jc][r] = t[r] * scale;
    }

    // online softmax over the 32 new columns
    float alpha[8], rs[8];
#pragma unroll
    for (int r = 0; r < 8; ++r) {
      const float mx = redmax16(fmaxf(s[0][r], s[1][r]));
      const float mn = fmaxf(mrun[r], mx);
      alpha[r] = __expf(mrun[r] - mn);
      mrun[r] = mn;
      const float p0 = __expf(s[0][r] - mn);
      const float p1 = __expf(s[1][r] - mn);
      s[0][r] = p0; s[1][r] = p1;
      rs[r] = redsum16(p0 + p1);
    }
#pragma unroll
    for (int r = 0; r < 8; ++r) lsum[r] = lsum[r] * alpha[r] + rs[r];
#pragma unroll
    for (int t = 0; t < 4; ++t)
#pragma unroll
      for (int r = 0; r < 8; ++r) O[t][r] *= alpha[r];

    // restage P: C-layout -> LDS -> A-layout (wave-private; DS in-order per wave)
#pragma unroll
    for (int jc = 0; jc < 2; ++jc)
#pragma unroll
      for (int r = 0; r < 8; ++r)
        plds[wave][r + lhi * 8][jc * 16 + lrow] = (bf16)s[jc][r];

    const int kk = lhi * 16;
    v16bf ap = ld_frag_a(&plds[wave][lrow][kk], &plds[wave][lrow][kk + 8]);

#pragma unroll
    for (int t = 0; t < 4; ++t) {
      const int d = t * 16 + lrow;
      v16bf bv = *(const v16bf*)(vb + (size_t)d * NCTX + j2 * 32 + lhi * 16);
      O[t] = wmma_bf16(ap, bv, O[t]);
    }
  }

  // normalize + store ctx [B, N, C] with col = h*64+d
#pragma unroll
  for (int t = 0; t < 4; ++t) {
#pragma unroll
    for (int r = 0; r < 8; ++r) {
      const int row = rblk + r + lhi * 8;
      const int d = t * 16 + lrow;
      const float v = O[t][r] / lsum[r];
      ctx[((size_t)(b * NCTX + row)) * DIM + h * HDIM + d] = (bf16)v;
    }
  }
}

// ---------------------------------------------------------------------------
// Host launcher
// ---------------------------------------------------------------------------
extern "C" void kernel_launch(void* const* d_in, const int* in_sizes, int n_in,
                              void* d_out, int out_size, void* d_ws, size_t ws_size,
                              hipStream_t stream) {
  (void)in_sizes; (void)n_in; (void)out_size; (void)ws_size;

  const float* x      = (const float*)d_in[0];
  const float* ln1_g  = (const float*)d_in[1];
  const float* ln1_b  = (const float*)d_in[2];
  const float* qkv_w  = (const float*)d_in[3];
  const float* proj_w = (const float*)d_in[4];
  const float* proj_b = (const float*)d_in[5];
  const float* ln2_g  = (const float*)d_in[6];
  const float* ln2_b  = (const float*)d_in[7];
  const float* fc1_w  = (const float*)d_in[8];
  const float* fc1_b  = (const float*)d_in[9];
  const float* fc2_w  = (const float*)d_in[10];
  const float* fc2_b  = (const float*)d_in[11];
  float* out = (float*)d_out;

  char* ws = (char*)d_ws;
  size_t off = 0;
  auto carve = [&](size_t bytes) -> char* {
    char* p = ws + off;
    off += (bytes + 255) & ~(size_t)255;
    return p;
  };
  bf16* wq   = (bf16*)carve((size_t)QKVN  * DIM    * 2);
  bf16* wp   = (bf16*)carve((size_t)DIM   * DIM    * 2);
  bf16* w1   = (bf16*)carve((size_t)HIDDEN* DIM    * 2);
  bf16* w2   = (bf16*)carve((size_t)DIM   * HIDDEN * 2);
  bf16* lno  = (bf16*)carve((size_t)ROWS  * DIM    * 2);   // ln1 out, reused for ln2 out
  bf16* big  = (bf16*)carve((size_t)ROWS  * HIDDEN * 2);   // qkv out, reused for FC1 out
  bf16* q    = (bf16*)carve((size_t)BATCH * HEADS * NCTX * HDIM * 2);
  bf16* k    = (bf16*)carve((size_t)BATCH * HEADS * NCTX * HDIM * 2);
  bf16* vt   = (bf16*)carve((size_t)BATCH * HEADS * NCTX * HDIM * 2);
  bf16* ctx  = (bf16*)carve((size_t)ROWS  * DIM    * 2);
  float* x2  = (float*)carve((size_t)ROWS * DIM    * 4);

  {
    int n;
    n = QKVN * DIM;    cvt_bf16_kernel<<<(n + 255) / 256, 256, 0, stream>>>(qkv_w,  wq, n);
    n = DIM * DIM;     cvt_bf16_kernel<<<(n + 255) / 256, 256, 0, stream>>>(proj_w, wp, n);
    n = HIDDEN * DIM;  cvt_bf16_kernel<<<(n + 255) / 256, 256, 0, stream>>>(fc1_w,  w1, n);
    n = DIM * HIDDEN;  cvt_bf16_kernel<<<(n + 255) / 256, 256, 0, stream>>>(fc2_w,  w2, n);
  }

  // LN1
  ln_kernel<<<ROWS, 256, 0, stream>>>(x, ln1_g, ln1_b, lno);

  // QKV GEMM -> bf16 [8192,2304]
  gemm_bf16_kernel<0><<<dim3(QKVN / 128, ROWS / 128), 256, 0, stream>>>(
      lno, wq, ROWS, QKVN, DIM, nullptr, nullptr, nullptr, big);

  // split q / k / v^T
  {
    const size_t n = (size_t)ROWS * QKVN;
    qkv_split_kernel<<<(unsigned)(n / 256), 256, 0, stream>>>(big, q, k, vt);
  }

  // flash attention -> ctx bf16
  attn_kernel<<<BATCH * HEADS * (NCTX / 128), 256, 0, stream>>>(q, k, vt, ctx);

  // proj + bias + residual(x) -> x2 fp32
  gemm_bf16_kernel<1><<<dim3(DIM / 128, ROWS / 128), 256, 0, stream>>>(
      ctx, wp, ROWS, DIM, DIM, proj_b, x, x2, nullptr);

  // LN2
  ln_kernel<<<ROWS, 256, 0, stream>>>(x2, ln2_g, ln2_b, lno);

  // FC1 + bias + GELU -> bf16 [8192,3072]
  gemm_bf16_kernel<2><<<dim3(HIDDEN / 128, ROWS / 128), 256, 0, stream>>>(
      lno, w1, ROWS, HIDDEN, DIM, fc1_b, nullptr, nullptr, big);

  // FC2 + bias + residual(x2) -> d_out fp32
  gemm_bf16_kernel<1><<<dim3(DIM / 128, ROWS / 128), 256, 0, stream>>>(
      big, w2, ROWS, DIM, HIDDEN, fc2_b, x2, out, nullptr);
}
